// MoELayerExercise_37391985279406
// MI455X (gfx1250) — matrix operand order repaired and checked by reading, not compiled, not run on hardware
//
#include <hip/hip_runtime.h>

#define N_TOK 2048
#define DDIM  1024
#define FDIM  2048
#define NEXP  8

typedef __attribute__((ext_vector_type(16))) __bf16 v16bf;
typedef __attribute__((ext_vector_type(8)))  float  v8f;

__device__ __forceinline__ unsigned short f2bf(float f) {
  unsigned u = __float_as_uint(f);
  u += 0x7FFFu + ((u >> 16) & 1u);      // round-to-nearest-even
  return (unsigned short)(u >> 16);
}

// ---------------- fp32 -> bf16, plain row-major (for x) ----------------
__global__ void cvt_x_kernel(const float* __restrict__ src,
                             unsigned short* __restrict__ dst, int n) {
  int i = (blockIdx.x * blockDim.x + threadIdx.x) * 4;
  if (i >= n) return;
  float4 f = *(const float4*)(src + i);
  unsigned lo = (unsigned)f2bf(f.x) | ((unsigned)f2bf(f.y) << 16);
  unsigned hi = (unsigned)f2bf(f.z) | ((unsigned)f2bf(f.w) << 16);
  uint2 o; o.x = lo; o.y = hi;
  *(uint2*)(dst + i) = o;
}

// ---- fp32 [R][ncols] -> bf16 K-pair-interleaved words: word[p*ncols+c] = {row 2p, row 2p+1} ----
__global__ void cvt_pair_kernel(const float* __restrict__ src,
                                unsigned int* __restrict__ dst,
                                int ncols, int nwords) {
  int w = blockIdx.x * blockDim.x + threadIdx.x;
  if (w >= nwords) return;
  int p = w / ncols;
  int c = w - p * ncols;
  float a = src[(size_t)(2 * p) * ncols + c];
  float b = src[(size_t)(2 * p + 1) * ncols + c];
  dst[w] = (unsigned)f2bf(a) | ((unsigned)f2bf(b) << 16);
}

// ---------------- router: one wave per token ----------------
__global__ __launch_bounds__(256)
void router_kernel(const float* __restrict__ x, const float* __restrict__ wg,
                   int* __restrict__ cnt, int* __restrict__ list,
                   float* __restrict__ listw) {
  int lane = threadIdx.x & 31;
  int wv   = threadIdx.x >> 5;
  int t    = blockIdx.x * 8 + wv;
  const float* xr = x + (size_t)t * DDIM;

  float acc[NEXP];
#pragma unroll
  for (int e = 0; e < NEXP; ++e) acc[e] = 0.0f;

  for (int i = 0; i < DDIM / 32; ++i) {
    int d = lane + i * 32;
    float xv = xr[d];
    const float4* g = (const float4*)(wg + (size_t)d * NEXP);
    float4 g0 = g[0], g1 = g[1];
    acc[0] += xv * g0.x; acc[1] += xv * g0.y;
    acc[2] += xv * g0.z; acc[3] += xv * g0.w;
    acc[4] += xv * g1.x; acc[5] += xv * g1.y;
    acc[6] += xv * g1.z; acc[7] += xv * g1.w;
  }
#pragma unroll
  for (int off = 16; off > 0; off >>= 1)
#pragma unroll
    for (int e = 0; e < NEXP; ++e) acc[e] += __shfl_xor(acc[e], off, 32);

  if (lane == 0) {
    float m = acc[0];
#pragma unroll
    for (int e = 1; e < NEXP; ++e) m = fmaxf(m, acc[e]);
    float p[NEXP];
#pragma unroll
    for (int e = 0; e < NEXP; ++e) p[e] = __expf(acc[e] - m);
    int i0 = 0; float b0 = p[0];
#pragma unroll
    for (int e = 1; e < NEXP; ++e) if (p[e] > b0) { b0 = p[e]; i0 = e; }
    int i1 = (i0 == 0) ? 1 : 0; float b1 = p[i1];
#pragma unroll
    for (int e = 0; e < NEXP; ++e)
      if (e != i0 && p[e] > b1) { b1 = p[e]; i1 = e; }
    float inv = 1.0f / (b0 + b1);     // renormalized top-2 (softmax denom cancels)
    int p0 = atomicAdd(cnt + i0, 1);
    list[i0 * N_TOK + p0]  = t;  listw[i0 * N_TOK + p0] = b0 * inv;
    int p1 = atomicAdd(cnt + i1, 1);
    list[i1 * N_TOK + p1]  = t;  listw[i1 * N_TOK + p1] = b1 * inv;
  }
}

// ---------------- expert FFN: 16 gathered tokens per block, 8 waves ----------------
__global__ __launch_bounds__(256)
void moe_ffn_kernel(const unsigned short* __restrict__ xb,  // [N][D] bf16
                    const unsigned int*   __restrict__ w1w, // [E][D/2][F] pair-words
                    const unsigned int*   __restrict__ w2w, // [E][F/2][D] pair-words
                    const int*   __restrict__ cnt,
                    const int*   __restrict__ list,
                    const float* __restrict__ listw,
                    float* __restrict__ out) {
  __shared__ unsigned short lds_x[16 * DDIM]; // 32 KB gathered activations
  __shared__ unsigned short lds_h[16 * 128];  // 4 KB h-chunk (relu'd, bf16)
  __shared__ int   s_tok[16];
  __shared__ float s_wt[16];

  const int e        = blockIdx.y;
  const int c        = cnt[e];
  const int tileBase = blockIdx.x * 16;
  if (tileBase >= c) return;                  // uniform exit

  const int tid  = threadIdx.x;
  const int lane = tid & 31;
  const int wv   = tid >> 5;
  const int half = lane >> 4;                 // lane half per WMMA layout
  const int l16  = lane & 15;
  const int kbA  = half * 16;                 // A: K offset within 32-chunk (u16)
  const int kbB  = half * 16;                 // B: K offset within 32-chunk

  if (tid < 16) {
    int idx = tileBase + tid;
    if (idx < c) { s_tok[tid] = list[e * N_TOK + idx]; s_wt[tid] = listw[e * N_TOK + idx]; }
    else         { s_tok[tid] = 0;                     s_wt[tid] = 0.0f; }
  }
  __syncthreads();

  { // async gather of 16 token rows (bf16) into LDS: 128 B per thread,
    // GLOBAL_LOAD_ASYNC_TO_LDS_B128 applies INST_OFFSET to both addresses.
    int m = tid >> 4, seg = tid & 15;
    unsigned long long gaddr =
        (unsigned long long)(xb + (size_t)s_tok[m] * DDIM + seg * 64);
    unsigned laddr = (unsigned)(size_t)(lds_x + m * DDIM + seg * 64);
#pragma unroll
    for (int i = 0; i < 8; ++i) {
      asm volatile("global_load_async_to_lds_b128 %0, %1, off offset:%2"
                   :: "v"(laddr), "v"(gaddr), "i"(i * 16) : "memory");
    }
    asm volatile("s_wait_asynccnt 0x0" ::: "memory");
  }
  __syncthreads();

  const unsigned int* w1e = w1w + (size_t)e * (DDIM / 2) * FDIM;
  const unsigned int* w2e = w2w + (size_t)e * (FDIM / 2) * DDIM;

  v8f oacc[8];
#pragma unroll
  for (int i = 0; i < 8; ++i) oacc[i] = v8f{0.f, 0.f, 0.f, 0.f, 0.f, 0.f, 0.f, 0.f};

  for (int fc = 0; fc < FDIM / 128; ++fc) {
    // ---- up-proj: this wave computes 16x16 tile of relu(x @ W1) ----
    const int fBase = fc * 128 + wv * 16;
    // warm L2->WGP path for this wave's W1 column block and upcoming W2 rows
    __builtin_prefetch((const void*)(w1e + (size_t)(kbB >> 1) * FDIM + fBase + l16), 0, 0);
    __builtin_prefetch((const void*)(w2e + (size_t)((fc * 128) >> 1) * DDIM + wv * 16 + l16), 0, 0);

    v8f hacc = v8f{0.f, 0.f, 0.f, 0.f, 0.f, 0.f, 0.f, 0.f};
#pragma unroll 4
    for (int ks = 0; ks < DDIM / 32; ++ks) {
      int k0 = ks * 32;
      union { uint4 q[2]; v16bf v; } A;
      const unsigned short* ap = lds_x + l16 * DDIM + k0 + kbA;
      A.q[0] = *(const uint4*)(ap);
      A.q[1] = *(const uint4*)(ap + 8);
      union { unsigned int w[8]; v16bf v; } B;
      const unsigned int* bp = w1e + (size_t)((k0 + kbB) >> 1) * FDIM + fBase + l16;
#pragma unroll
      for (int v = 0; v < 8; ++v) B.w[v] = bp[(size_t)v * FDIM];
      hacc = __builtin_amdgcn_wmma_f32_16x16x32_bf16(false, A.v, false, B.v,
                                                     (short)0, hacc, false, false);
    }
    __syncthreads();  // prior chunk's readers done before overwriting lds_h
#pragma unroll
    for (int v = 0; v < 8; ++v) {
      int M = v + half * 8;
      float val = hacc[v];
      lds_h[M * 128 + wv * 16 + l16] = f2bf(val > 0.f ? val : 0.f);
    }
    __syncthreads();

    // ---- down-proj partial: accumulate h-chunk @ W2 into persistent out tiles ----
#pragma unroll
    for (int ks = 0; ks < 4; ++ks) {
      int k0 = ks * 32;
      union { uint4 q[2]; v16bf v; } A;
      const unsigned short* ap = lds_h + l16 * 128 + k0 + kbA;
      A.q[0] = *(const uint4*)(ap);
      A.q[1] = *(const uint4*)(ap + 8);
      int kglob = fc * 128 + k0 + kbB;
#pragma unroll
      for (int dt = 0; dt < 8; ++dt) {
        int dBase = (dt * 8 + wv) * 16;
        union { unsigned int w[8]; v16bf v; } B;
        const unsigned int* bp = w2e + (size_t)(kglob >> 1) * DDIM + dBase + l16;
#pragma unroll
        for (int v = 0; v < 8; ++v) B.w[v] = bp[(size_t)v * DDIM];
        oacc[dt] = __builtin_amdgcn_wmma_f32_16x16x32_bf16(false, A.v, false, B.v,
                                                           (short)0, oacc[dt], false, false);
      }
    }
  }

  // ---- scaled scatter-add into output ----
#pragma unroll
  for (int dt = 0; dt < 8; ++dt) {
    int dBase = (dt * 8 + wv) * 16;
#pragma unroll
    for (int v = 0; v < 8; ++v) {
      int M = v + half * 8;
      atomicAdd(out + (size_t)s_tok[M] * DDIM + dBase + l16, oacc[dt][v] * s_wt[M]);
    }
  }
}

extern "C" void kernel_launch(void* const* d_in, const int* in_sizes, int n_in,
                              void* d_out, int out_size, void* d_ws, size_t ws_size,
                              hipStream_t stream) {
  const float* x  = (const float*)d_in[0];
  const float* Wg = (const float*)d_in[1];
  const float* W1 = (const float*)d_in[2];
  const float* W2 = (const float*)d_in[3];
  float* out = (float*)d_out;

  char* ws = (char*)d_ws;
  size_t o = 0;
  unsigned short* xb  = (unsigned short*)(ws + o); o += (size_t)N_TOK * DDIM * 2;          // 4 MB
  unsigned int*   w1w = (unsigned int*)  (ws + o); o += (size_t)NEXP * (DDIM/2) * FDIM * 4; // 32 MB
  unsigned int*   w2w = (unsigned int*)  (ws + o); o += (size_t)NEXP * (FDIM/2) * DDIM * 4; // 32 MB
  int*   cnt   = (int*)  (ws + o); o += 256;
  int*   list  = (int*)  (ws + o); o += (size_t)NEXP * N_TOK * 4;
  float* listw = (float*)(ws + o);

  hipMemsetAsync(d_out, 0, (size_t)out_size * sizeof(float), stream);
  hipMemsetAsync(cnt, 0, NEXP * sizeof(int), stream);

  cvt_x_kernel<<<(N_TOK * DDIM / 4 + 255) / 256, 256, 0, stream>>>(x, xb, N_TOK * DDIM);
  {
    int nw = NEXP * (DDIM / 2) * FDIM;
    cvt_pair_kernel<<<(nw + 255) / 256, 256, 0, stream>>>(W1, w1w, FDIM, nw);
  }
  {
    int nw = NEXP * (FDIM / 2) * DDIM;
    cvt_pair_kernel<<<(nw + 255) / 256, 256, 0, stream>>>(W2, w2w, DDIM, nw);
  }
  router_kernel<<<N_TOK / 8, 256, 0, stream>>>(x, Wg, cnt, list, listw);
  moe_ffn_kernel<<<dim3(N_TOK / 16, NEXP), 256, 0, stream>>>(xb, w1w, w2w,
                                                             cnt, list, listw, out);
}
